// FullyContextualEmbedding_58334245814652
// MI455X (gfx1250) — matrix operand order repaired and checked by reading, not compiled, not run on hardware
//
#include <hip/hip_runtime.h>
#include <hip/hip_bf16.h>
#include <math.h>

// ---------------------------------------------------------------------------
// FullyContextualEmbedding on MI455X (gfx1250, wave32, WMMA)
//
// B=4096, D=1024, K(mem)=64, 64 recurrent steps.
// Dominant cost: per-step gates GEMM  x[B,3D] @ Wcat^T[4D,3D]  (103 GFLOP/step,
// 6.6 TFLOP total) -> compute bound (weights L2-resident: 24MB << 192MB L2;
// total HBM traffic ~6GB << 23.3TB/s * runtime). Therefore bf16 WMMA
// (v_wmma_f32_16x16x32_bf16) with fp32 accumulate; cell state c and the
// pointwise LSTM update stay fp32.
//
// CDNA5-specific paths used:
//   - v_wmma_f32_16x16x32_bf16 (all GEMMs)
//   - global_load_async_to_lds_b128 + s_wait_asynccnt (double-buffered W tiles)
// ---------------------------------------------------------------------------

typedef __attribute__((ext_vector_type(16))) __bf16 v16bf;
typedef __attribute__((ext_vector_type(8)))  __bf16 v8bf;
typedef __attribute__((ext_vector_type(8)))  float  v8f;

union ABu { v16bf v; v8bf h[2]; };

#define BB    4096
#define DD    1024
#define KM    64
#define XK    3072     // concat K dim: [f(1024) | r(1024) | h(1024)]
#define FOURD 4096

__device__ __forceinline__ float sigm(float v) { return 1.0f / (1.0f + __expf(-v)); }

__device__ __forceinline__ v8f wmma_bf16(const ABu& a, const ABu& b, v8f c) {
    return __builtin_amdgcn_wmma_f32_16x16x32_bf16(
        /*neg_a=*/false, a.v, /*neg_b=*/false, b.v,
        /*c_mod=*/(short)0, c, /*reuse_a=*/false, /*reuse_b=*/false);
}

// Async memory->LDS copy of 16 bytes (ASYNCcnt-tracked, no VGPR round trip).
// ldsoff = workgroup-relative LDS byte offset (low 32 bits of generic pointer).
__device__ __forceinline__ void async_copy_b128(unsigned ldsoff, const void* gptr) {
    asm volatile("global_load_async_to_lds_b128 %0, %1, off"
                 :: "v"(ldsoff), "v"(gptr) : "memory");
}
__device__ __forceinline__ void wait_asynccnt0() {
    asm volatile("s_wait_asynccnt 0x0" ::: "memory");
}

// ---------------------------------------------------------------------------
// Prep kernels (run once; outside the recurrent loop)
// ---------------------------------------------------------------------------

// Wcat[n][k] (bf16, [4096][3072], N-major => contiguous-K rows for B fragments)
__global__ __launch_bounds__(256) void init_wcat_kernel(
    const float* __restrict__ wih, const float* __restrict__ whh,
    __bf16* __restrict__ wcat) {
    size_t i = (size_t)blockIdx.x * 256 + threadIdx.x;   // 4096*3072 total
    int n = (int)(i / XK);
    int k = (int)(i % XK);
    float v = (k < 2048) ? wih[(size_t)n * 2048 + k]
                         : whh[(size_t)n * 1024 + (k - 2048)];
    wcat[i] = (__bf16)v;
}

// x0/x1: f part = f, h part = f (h0 = f per the scan carry), r part = 0; c = 0
__global__ __launch_bounds__(256) void init_state_kernel(
    const float* __restrict__ f, __bf16* __restrict__ x0,
    __bf16* __restrict__ x1, float* __restrict__ c) {
    size_t i = (size_t)blockIdx.x * 256 + threadIdx.x;   // 4096*1024 total
    int b = (int)(i >> 10);
    int d = (int)(i & 1023);
    float v = f[i];
    __bf16 bv = (__bf16)v;
    size_t row = (size_t)b * XK;
    x0[row + d]        = bv;          x1[row + d]        = bv;      // f part
    x0[row + 2048 + d] = bv;          x1[row + 2048 + d] = bv;      // h0 = f
    x0[row + 1024 + d] = (__bf16)0.f; x1[row + 1024 + d] = (__bf16)0.f;
    c[i] = 0.0f;
}

// gbf[kmem][d] = bf16(G)  (N(mem)-major, contiguous d  -> logits B frags)
// gt [d][kmem] = bf16(G^T)(N(feat)-major, contiguous mem -> readout B frags)
__global__ __launch_bounds__(256) void init_g_kernel(
    const float* __restrict__ G, __bf16* __restrict__ gbf,
    __bf16* __restrict__ gt) {
    int i = blockIdx.x * 256 + threadIdx.x;              // 64*1024 total
    int k = i >> 10;          // memory index
    int d = i & 1023;         // feature index
    float v = G[i];
    gbf[i] = (__bf16)v;
    gt[(size_t)d * KM + k] = (__bf16)v;
}

__global__ __launch_bounds__(256) void init_bias_kernel(
    const float* __restrict__ bih, const float* __restrict__ bhh,
    float* __restrict__ bias) {
    int i = blockIdx.x * 256 + threadIdx.x;              // 4096 total
    bias[i] = bih[i] + bhh[i];
}

// ---------------------------------------------------------------------------
// Attention + readout:  a = softmax(h G^T),  r = a G   (writes r into x)
// grid = 32 blocks, 256 threads (8 waves); wave w handles 16 rows of B.
// ---------------------------------------------------------------------------
__global__ __launch_bounds__(256) void attn_kernel(
    __bf16* __restrict__ x,               // [B][XK]; h at +2048, r at +1024
    const __bf16* __restrict__ gbf,       // [64][1024]
    const __bf16* __restrict__ gt) {      // [1024][64]
    __shared__ float  lw[8 * 16 * KM];    // per-wave logits (32 KB)
    __shared__ __bf16 al[8 * 16 * KM];    // per-wave softmax probs (16 KB)

    const int tid  = threadIdx.x;
    const int lane = tid & 31;
    const int w    = tid >> 5;
    const int nl   = lane & 15;
    const int half = lane >> 4;
    const int m0   = blockIdx.x * 128 + w * 16;

    // ---- logits = h @ G^T : 4 accumulators cover all 64 memories ----
    const __bf16* hrow = x + (size_t)(m0 + nl) * XK + 2048;
    v8f acc[4];
#pragma unroll
    for (int t = 0; t < 4; ++t)
#pragma unroll
        for (int e = 0; e < 8; ++e) acc[t][e] = 0.0f;

    for (int kk = 0; kk < 32; ++kk) {            // K = 1024, 32 per WMMA
        ABu a;
        a.h[0] = *(const v8bf*)(hrow + kk * 32 + half * 8);
        a.h[1] = *(const v8bf*)(hrow + kk * 32 + half * 8 + 16);
#pragma unroll
        for (int t = 0; t < 4; ++t) {
            ABu bf;
            const __bf16* bp = gbf + (size_t)(t * 16 + nl) * DD + kk * 32 + half * 8;
            bf.h[0] = *(const v8bf*)(bp);
            bf.h[1] = *(const v8bf*)(bp + 16);
            acc[t] = wmma_bf16(a, bf, acc[t]);
        }
    }

    // dump logits: element e of acc[t] = (m = half*8+e, mem = 16t+nl)
    float* lww = lw + w * 16 * KM;
#pragma unroll
    for (int t = 0; t < 4; ++t)
#pragma unroll
        for (int e = 0; e < 8; ++e)
            lww[(half * 8 + e) * KM + t * 16 + nl] = acc[t][e];
    __syncthreads();

    // ---- softmax: lanes 0..15 of each wave own one row each ----
    __bf16* alw = al + w * 16 * KM;
    if (lane < 16) {
        float row[KM];
        float mx = -3.0e38f;
#pragma unroll
        for (int k = 0; k < KM; ++k) { row[k] = lww[lane * KM + k]; mx = fmaxf(mx, row[k]); }
        float s = 0.0f;
#pragma unroll
        for (int k = 0; k < KM; ++k) { row[k] = __expf(row[k] - mx); s += row[k]; }
        float inv = 1.0f / s;
#pragma unroll
        for (int k = 0; k < KM; ++k) alw[lane * KM + k] = (__bf16)(row[k] * inv);
    }
    __syncthreads();

    // ---- readout r = a @ G : 64 feature tiles, K(mem)=64 => 2 WMMAs each ----
    for (int nt = 0; nt < 64; ++nt) {
        v8f r;
#pragma unroll
        for (int e = 0; e < 8; ++e) r[e] = 0.0f;
#pragma unroll
        for (int kk = 0; kk < 2; ++kk) {
            ABu a;
            const __bf16* ap = alw + nl * KM + kk * 32 + half * 8;
            a.h[0] = *(const v8bf*)(ap);
            a.h[1] = *(const v8bf*)(ap + 16);
            ABu bf;
            const __bf16* bp = gt + (size_t)(nt * 16 + nl) * KM + kk * 32 + half * 8;
            bf.h[0] = *(const v8bf*)(bp);
            bf.h[1] = *(const v8bf*)(bp + 16);
            r = wmma_bf16(a, bf, r);
        }
#pragma unroll
        for (int e = 0; e < 8; ++e) {
            int b = m0 + half * 8 + e;
            x[(size_t)b * XK + 1024 + nt * 16 + nl] = (__bf16)r[e];
        }
    }
}

// ---------------------------------------------------------------------------
// Gates GEMM + LSTM cell update.
// grid = (16, 32), 256 threads (8 waves). Wave w computes the 32x32 tile of
// all FOUR gates at rows [m0,m0+32), cols [n0,n0+32): 16 accumulators; per
// K-step one pair of A fragments is reused against 8 B fragments staged in a
// double-buffered LDS tile filled with global_load_async_to_lds_b128 and
// synchronized with s_wait_asynccnt (one barrier per K-step).
// Epilogue: fp32 sigmoid/tanh cell update; h_new -> x_next (bf16), and
// (last step only) fp32 h -> d_out.
// ---------------------------------------------------------------------------
#define WTILE (4 * 32 * 32)   // 4 gates x 32 n x 32 k = 4096 bf16 = 8 KB

__global__ __launch_bounds__(256) void gates_kernel(
    const __bf16* __restrict__ x,         // x_cur [B][XK]
    const __bf16* __restrict__ wcat,      // [4096][3072]
    const float* __restrict__ bias,       // [4096]
    const float* __restrict__ f_in,       // [B][D] fp32 (skip connection)
    float* __restrict__ c,                // [B][D] fp32 cell state
    __bf16* __restrict__ x_next,          // h part (+2048) written here
    float* __restrict__ h_out) {          // nullptr except on last step
    __shared__ __bf16 wl[2][WTILE];       // double-buffered W tile (16 KB)

    const int tid  = threadIdx.x;
    const int lane = tid & 31;
    const int w    = tid >> 5;
    const int nl   = lane & 15;
    const int half = lane >> 4;
    const int m0   = blockIdx.x * 256 + w * 32;
    const int n0   = blockIdx.y * 32;

    v8f acc[2][2][4];                     // [m-subtile][n-subtile][gate]
#pragma unroll
    for (int mt = 0; mt < 2; ++mt)
#pragma unroll
        for (int nt = 0; nt < 2; ++nt)
#pragma unroll
            for (int g = 0; g < 4; ++g)
#pragma unroll
                for (int e = 0; e < 8; ++e) acc[mt][nt][g][e] = 0.0f;

    const __bf16* xrow0 = x + (size_t)(m0 + nl) * XK;
    const __bf16* xrow1 = xrow0 + (size_t)16 * XK;

    // Tile-fill mapping: 512 x 16B chunks per buffer, 2 per thread.
    // chunk cid -> row r=cid>>2 (gate g=r>>5, n=r&31), k-chunk (cid&3)*8.
    const unsigned ldsbase = (unsigned)(uintptr_t)(void*)&wl[0][0];
    const int cid0 = tid;            // rows 0..63   (gates 0,1)
    const int cid1 = tid + 256;      // rows 64..127 (gates 2,3)
    const int r0 = cid0 >> 2, r1 = cid1 >> 2;
    const __bf16* gsrc0 =
        wcat + (size_t)((r0 >> 5) * DD + n0 + (r0 & 31)) * XK + (cid0 & 3) * 8;
    const __bf16* gsrc1 =
        wcat + (size_t)((r1 >> 5) * DD + n0 + (r1 & 31)) * XK + (cid1 & 3) * 8;
    const unsigned dst0 = ldsbase + (unsigned)cid0 * 16u;
    const unsigned dst1 = ldsbase + (unsigned)cid1 * 16u;

    // prologue: async-fill buffer 0 with tile k=0
    async_copy_b128(dst0, gsrc0);
    async_copy_b128(dst1, gsrc1);

    for (int kk = 0; kk < 96; ++kk) {           // K = 3072, 32 per WMMA
        const int k0 = kk * 32;
        wait_asynccnt0();                       // my tile-k chunks landed in LDS
        __syncthreads();                        // whole tile visible; prev reads done

        if (kk + 1 < 96) {                      // prefetch tile k+1 into other buffer
            const unsigned boff = (unsigned)(((kk + 1) & 1) * WTILE * 2);
            const int k1 = (kk + 1) * 32;
            async_copy_b128(dst0 + boff, gsrc0 + k1);
            async_copy_b128(dst1 + boff, gsrc1 + k1);
        }

        ABu a[2];
        a[0].h[0] = *(const v8bf*)(xrow0 + k0 + half * 8);
        a[0].h[1] = *(const v8bf*)(xrow0 + k0 + half * 8 + 16);
        a[1].h[0] = *(const v8bf*)(xrow1 + k0 + half * 8);
        a[1].h[1] = *(const v8bf*)(xrow1 + k0 + half * 8 + 16);

        const __bf16* wb = &wl[kk & 1][0];
#pragma unroll
        for (int g = 0; g < 4; ++g)
#pragma unroll
            for (int nt = 0; nt < 2; ++nt) {
                ABu bf;
                const __bf16* bp = wb + ((g * 32 + nt * 16 + nl) * 32 + half * 8);
                bf.h[0] = *(const v8bf*)(bp);
                bf.h[1] = *(const v8bf*)(bp + 16);
                acc[0][nt][g] = wmma_bf16(a[0], bf, acc[0][nt][g]);
                acc[1][nt][g] = wmma_bf16(a[1], bf, acc[1][nt][g]);
            }
    }

    // ---- epilogue: LSTM cell update in fp32 ----
#pragma unroll
    for (int mt = 0; mt < 2; ++mt)
#pragma unroll
        for (int nt = 0; nt < 2; ++nt) {
            const int d = n0 + nt * 16 + nl;
            const float b_i = bias[d];
            const float b_f = bias[DD + d];
            const float b_g = bias[2 * DD + d];
            const float b_o = bias[3 * DD + d];
#pragma unroll
            for (int e = 0; e < 8; ++e) {
                const int b = m0 + mt * 16 + half * 8 + e;
                const size_t idx = (size_t)b * DD + d;
                const float ig  = acc[mt][nt][0][e] + b_i;
                const float fg2 = acc[mt][nt][1][e] + b_f;
                const float gg  = acc[mt][nt][2][e] + b_g;
                const float og  = acc[mt][nt][3][e] + b_o;
                const float cn = sigm(fg2) * c[idx] + sigm(ig) * tanhf(gg);
                const float hn = sigm(og) * tanhf(cn) + f_in[idx]; // skip: h=h+f
                c[idx] = cn;
                x_next[(size_t)b * XK + 2048 + d] = (__bf16)hn;
                if (h_out) h_out[idx] = hn;
            }
        }
}

// ---------------------------------------------------------------------------
// Host-side orchestration (graph-capture safe: only kernel launches on stream)
// ---------------------------------------------------------------------------
extern "C" void kernel_launch(void* const* d_in, const int* in_sizes, int n_in,
                              void* d_out, int out_size, void* d_ws, size_t ws_size,
                              hipStream_t stream) {
    const float* f   = (const float*)d_in[0];   // [4096,1024]
    const float* G   = (const float*)d_in[1];   // [64,1024]
    const float* Wih = (const float*)d_in[2];   // [4096,2048]
    const float* Whh = (const float*)d_in[3];   // [4096,1024]
    const float* bih = (const float*)d_in[4];   // [4096]
    const float* bhh = (const float*)d_in[5];   // [4096]
    float* out = (float*)d_out;                 // [4096,1024] fp32

    // workspace layout (~88.3 MB)
    char* ws = (char*)d_ws;
    __bf16* x0   = (__bf16*)(ws);                      // 25,165,824 B
    __bf16* x1   = (__bf16*)(ws + 25165824ull);        // 25,165,824 B
    float*  c    = (float*) (ws + 50331648ull);        // 16,777,216 B
    __bf16* wcat = (__bf16*)(ws + 67108864ull);        // 25,165,824 B
    __bf16* gbf  = (__bf16*)(ws + 92274688ull);        //    131,072 B
    __bf16* gt   = (__bf16*)(ws + 92405760ull);        //    131,072 B
    float*  bias = (float*) (ws + 92536832ull);        //     16,384 B

    init_wcat_kernel <<<49152, 256, 0, stream>>>(Wih, Whh, wcat);
    init_state_kernel<<<16384, 256, 0, stream>>>(f, x0, x1, c);
    init_g_kernel    <<<256,   256, 0, stream>>>(G, gbf, gt);
    init_bias_kernel <<<16,    256, 0, stream>>>(bih, bhh, bias);

    for (int s = 0; s < 64; ++s) {
        __bf16* xc = (s & 1) ? x1 : x0;
        __bf16* xn = (s & 1) ? x0 : x1;
        attn_kernel <<<32, 256, 0, stream>>>(xc, gbf, gt);
        gates_kernel<<<dim3(16, 32), 256, 0, stream>>>(
            xc, wcat, bias, f, c, xn, (s == 63) ? out : nullptr);
    }
}
